// LSTM_Decoder_60447369724230
// MI455X (gfx1250) — compile-verified, matrix operand order checked by reading, not run on hardware
//
#include <hip/hip_runtime.h>
#include <hip/hip_bf16.h>

// ---------------------------------------------------------------------------
// LSTM decoder for MI455X (gfx1250): bf16 WMMA GEMMs with fp32 accumulate.
//   T=64, B=32, V=32000, E=H=512.
// ---------------------------------------------------------------------------

typedef __attribute__((ext_vector_type(16))) __bf16 v16bf;
typedef __attribute__((ext_vector_type(8)))  float  v8f;
typedef __attribute__((ext_vector_type(4)))  int    v4i_;

union ABTile {
    v16bf          v;
    unsigned short s[16];
    uint4          q[2];
};

#if __has_builtin(__builtin_amdgcn_global_load_async_to_lds_b128) && \
    __has_builtin(__builtin_amdgcn_s_wait_asynccnt)
#define HAVE_ASYNC_LDS 1
#else
#define HAVE_ASYNC_LDS 0
#endif

// 16-byte global -> LDS copy; async (ASYNCcnt-tracked) when available.
static __device__ __forceinline__ void copy16_to_lds(unsigned short* lds_dst,
                                                     const unsigned short* gsrc) {
#if HAVE_ASYNC_LDS
    __builtin_amdgcn_global_load_async_to_lds_b128(
        (__attribute__((address_space(1))) v4i_*)gsrc,
        (__attribute__((address_space(3))) v4i_*)lds_dst, 0, 0);
#else
    *(uint4*)lds_dst = *(const uint4*)gsrc;
#endif
}

static __device__ __forceinline__ void lds_copy_join() {
#if HAVE_ASYNC_LDS
    __builtin_amdgcn_s_wait_asynccnt(0);
#endif
    __syncthreads();
}

// round-to-nearest-even fp32 -> bf16
static __device__ __forceinline__ unsigned short f2bf(float f) {
    unsigned int u = __float_as_uint(f);
    u += 0x7FFFu + ((u >> 16) & 1u);
    return (unsigned short)(u >> 16);
}

static __device__ __forceinline__ float sigmoidf_(float x) {
    return 1.0f / (1.0f + __expf(-x));
}

// ---------------------------------------------------------------------------
// One-time conversion kernels
// ---------------------------------------------------------------------------

// W_fc [32000,512] fp32 -> bf16 (33 MB, L2-resident afterwards)
__global__ void cvt_wfc_kernel(const float* __restrict__ w,
                               unsigned short* __restrict__ o, int n) {
    int i = blockIdx.x * blockDim.x + threadIdx.x;
    if (i < n) o[i] = f2bf(w[i]);
}

// wcomb[j][0:512] = W_ih[j][:], wcomb[j][512:1024] = W_hh[j][:]  (bf16)
__global__ void cvt_wcomb_kernel(const float* __restrict__ wih,
                                 const float* __restrict__ whh,
                                 unsigned short* __restrict__ o) {
    int i = blockIdx.x * blockDim.x + threadIdx.x;   // 2048*1024 threads
    int j = i >> 10;
    int k = i & 1023;
    float v = (k < 512) ? wih[j * 512 + k] : whh[j * 512 + (k - 512)];
    o[i] = f2bf(v);
}

// ---------------------------------------------------------------------------
// Per-step A-panel build: apanel[b][0:512]=bf16(emb[tok_b]), [512:1024]=bf16(h)
//   32*1024 threads. Hoists ALL fp32->bf16 conversion out of the GEMM loop.
// ---------------------------------------------------------------------------
__global__ __launch_bounds__(256)
void embed_gather_kernel(const int* __restrict__ tokens, int t,
                         const float* __restrict__ emb,
                         const float* __restrict__ hstate,
                         unsigned short* __restrict__ apanel) {
    int idx = blockIdx.x * blockDim.x + threadIdx.x;   // [0, 32768)
    int b = idx >> 10;
    int k = idx & 1023;
    float v;
    if (k < 512) {
        int tok = tokens[t * 32 + b];
        v = emb[(size_t)tok * 512 + k];
    } else {
        v = hstate[(size_t)b * 512 + (k - 512)];
    }
    apanel[idx] = f2bf(v);
}

// ---------------------------------------------------------------------------
// Per-step gates GEMM: gates[32,2048] = apanel[32,1024] @ wcomb^T  (no bias)
//   grid = (2 Mtiles, 16), block = 256 (8 waves); one 16x16 tile per wave.
//   A M-tile (16x1024 bf16) staged in LDS once per block, shared by 8 waves.
// ---------------------------------------------------------------------------
#define GATES_APAD 1032   // row stride in bf16 units (pad: 8 bf16 = 4 dwords)

__global__ __launch_bounds__(256)
void gates_kernel(const unsigned short* __restrict__ apanel,
                  const unsigned short* __restrict__ wcomb,
                  float* __restrict__ gates) {
    __shared__ unsigned short lA[16 * GATES_APAD];

    const int lane   = threadIdx.x & 31;
    const int wave   = threadIdx.x >> 5;
    const int lane16 = lane & 15;
    const int hi     = lane >> 4;
    const int mtile  = blockIdx.x;

    // cooperative stage: 16 rows x 1024 bf16 = 2048 x 16B chunks, 8 per thread
    #pragma unroll
    for (int c = 0; c < 8; ++c) {
        int chunk = threadIdx.x * 8 + c;     // [0, 2048)
        int row = chunk >> 7;
        int col = (chunk & 127) * 8;
        copy16_to_lds(lA + row * GATES_APAD + col,
                      apanel + (size_t)(mtile * 16 + row) * 1024 + col);
    }
    lds_copy_join();

    // B-matrix column (N = gate unit index)
    const int ntile = blockIdx.y * 8 + wave;
    const int j = ntile * 16 + lane16;
    const unsigned short* brow = wcomb + (size_t)j * 1024;
    const unsigned short* arow = lA + lane16 * GATES_APAD;

    v8f acc = {};
    #pragma unroll 4
    for (int kk = 0; kk < 32; ++kk) {
        const int ka = kk * 32 + 8 * hi;
        ABTile a;                                   // ds_load_b128 x2
        a.q[0] = *(const uint4*)(arow + ka);
        a.q[1] = *(const uint4*)(arow + ka + 16);
        ABTile bt;                                  // global_load_b128 x2
        bt.q[0] = *(const uint4*)(brow + ka);
        bt.q[1] = *(const uint4*)(brow + ka + 16);
        acc = __builtin_amdgcn_wmma_f32_16x16x32_bf16(
                  false, a.v, false, bt.v, (short)0, acc, false, false);
    }

    // C/D layout: lane -> N = lane16; VGPR r -> M = r + 8*hi
    #pragma unroll
    for (int r = 0; r < 8; ++r) {
        int m = mtile * 16 + r + 8 * hi;
        gates[(size_t)m * 2048 + j] = acc[r];
    }
}

// ---------------------------------------------------------------------------
// Per-step elementwise LSTM cell update + pad mask + bf16 h_tmp dump.
// ---------------------------------------------------------------------------
__global__ __launch_bounds__(256)
void lstm_update_kernel(const float* __restrict__ gates,
                        const float* __restrict__ b_ih,
                        const float* __restrict__ b_hh,
                        const int* __restrict__ tokens, int t,
                        float* __restrict__ hstate,
                        float* __restrict__ cstate,
                        unsigned short* __restrict__ hbf) {
    int idx = blockIdx.x * blockDim.x + threadIdx.x;   // [0, 16384)
    int b = idx >> 9;
    int k = idx & 511;

    const float* g = gates + (size_t)b * 2048;
    float gi = g[k]        + b_ih[k]        + b_hh[k];
    float gf = g[512 + k]  + b_ih[512 + k]  + b_hh[512 + k];
    float gg = g[1024 + k] + b_ih[1024 + k] + b_hh[1024 + k];
    float go = g[1536 + k] + b_ih[1536 + k] + b_hh[1536 + k];

    float i_ = sigmoidf_(gi);
    float f_ = sigmoidf_(gf);
    float gt = tanhf(gg);
    float o_ = sigmoidf_(go);

    float c     = cstate[idx];
    float c_tmp = f_ * c + i_ * gt;
    float h_tmp = o_ * tanhf(c_tmp);

    // logits see the *unmasked* tentative hidden
    hbf[((size_t)t * 32 + b) * 512 + k] = f2bf(h_tmp);

    if (tokens[t * 32 + b] != 0) {   // PAD == 0
        hstate[idx] = h_tmp;
        cstate[idx] = c_tmp;
    }
}

// ---------------------------------------------------------------------------
// Batched projection: out[2048,32000] = hbf[2048,512] @ W_fc^T + b_fc
//   grid = (128 Mtiles, 63), block = 256 (8 waves).
//   A M-tile (16x512 bf16) staged in LDS; each wave does 4 N-tiles per A tile.
// ---------------------------------------------------------------------------
#define PROJ_APAD 520     // row stride in bf16 units

__global__ __launch_bounds__(256)
void proj_kernel(const unsigned short* __restrict__ hbf,
                 const unsigned short* __restrict__ wfc,
                 const float* __restrict__ b_fc,
                 float* __restrict__ out) {
    __shared__ unsigned short lA[16 * PROJ_APAD];

    const int lane   = threadIdx.x & 31;
    const int wave   = threadIdx.x >> 5;
    const int lane16 = lane & 15;
    const int hi     = lane >> 4;
    const int mtile  = blockIdx.x;

    // cooperative stage: 16 rows x 512 bf16 = 1024 x 16B chunks, 4 per thread
    #pragma unroll
    for (int c = 0; c < 4; ++c) {
        int chunk = threadIdx.x * 4 + c;     // [0, 1024)
        int row = chunk >> 6;
        int col = (chunk & 63) * 8;
        copy16_to_lds(lA + row * PROJ_APAD + col,
                      hbf + (size_t)(mtile * 16 + row) * 512 + col);
    }
    lds_copy_join();

    const int ntbase = (blockIdx.y * 8 + wave) * 4;   // first of 4 N-tiles
    const unsigned short* arow = lA + lane16 * PROJ_APAD;

    const unsigned short* brow[4];
    bool valid[4];
    #pragma unroll
    for (int u = 0; u < 4; ++u) {
        int nt = ntbase + u;
        valid[u] = (nt < 2000);                       // 32000/16 = 2000 tiles
        brow[u] = wfc + (size_t)((valid[u] ? nt : 0) * 16 + lane16) * 512;
    }

    v8f acc[4] = {};
    for (int kk = 0; kk < 16; ++kk) {
        const int ka = kk * 32 + 8 * hi;
        ABTile a;                                     // ds_load_b128 x2
        a.q[0] = *(const uint4*)(arow + ka);
        a.q[1] = *(const uint4*)(arow + ka + 16);
        #pragma unroll
        for (int u = 0; u < 4; ++u) {
            if (!valid[u]) continue;                  // wave-uniform branch
            ABTile bt;
            bt.q[0] = *(const uint4*)(brow[u] + ka);
            bt.q[1] = *(const uint4*)(brow[u] + ka + 16);
            if (kk < 15)                              // prefetch next k-step
                __builtin_prefetch(brow[u] + ka + 32, 0, 1);
            acc[u] = __builtin_amdgcn_wmma_f32_16x16x32_bf16(
                         false, a.v, false, bt.v, (short)0, acc[u], false, false);
        }
    }

    #pragma unroll
    for (int u = 0; u < 4; ++u) {
        if (!valid[u]) continue;
        int nj = (ntbase + u) * 16 + lane16;
        float bias = b_fc[nj];
        #pragma unroll
        for (int r = 0; r < 8; ++r) {
            int m = mtile * 16 + r + 8 * hi;
            out[(size_t)m * 32000 + nj] = acc[u][r] + bias;
        }
    }
}

// ---------------------------------------------------------------------------
// Host-side launcher
// ---------------------------------------------------------------------------
extern "C" void kernel_launch(void* const* d_in, const int* in_sizes, int n_in,
                              void* d_out, int out_size, void* d_ws, size_t ws_size,
                              hipStream_t stream) {
    const int*   tokens = (const int*)  d_in[0];   // [64,32]
    const float* emb    = (const float*)d_in[1];   // [32000,512]
    const float* W_ih   = (const float*)d_in[2];   // [2048,512]
    const float* W_hh   = (const float*)d_in[3];   // [2048,512]
    const float* b_ih   = (const float*)d_in[4];   // [2048]
    const float* b_hh   = (const float*)d_in[5];   // [2048]
    const float* W_fc   = (const float*)d_in[6];   // [32000,512]
    const float* b_fc   = (const float*)d_in[7];   // [32000]
    const float* h0     = (const float*)d_in[8];   // [32,512]
    const float* c0     = (const float*)d_in[9];   // [32,512]
    float* out = (float*)d_out;                    // [64,32,32000]

    char* ws = (char*)d_ws;
    float*          hstate = (float*)(ws + 0);                        //  64 KB
    float*          cstate = (float*)(ws + (size_t)65536);            //  64 KB
    float*          gates  = (float*)(ws + (size_t)131072);           // 256 KB
    unsigned short* apanel = (unsigned short*)(ws + (size_t)393216);  //  64 KB
    unsigned short* hbf    = (unsigned short*)(ws + (size_t)458752);  //   2 MB
    unsigned short* wcomb  = (unsigned short*)(ws + (size_t)2555904); //   4 MB
    unsigned short* wfcb   = (unsigned short*)(ws + (size_t)6750208); //  33 MB

    // init recurrent state (graph-capture-safe async d2d copies)
    (void)hipMemcpyAsync(hstate, h0, 32 * 512 * sizeof(float),
                         hipMemcpyDeviceToDevice, stream);
    (void)hipMemcpyAsync(cstate, c0, 32 * 512 * sizeof(float),
                         hipMemcpyDeviceToDevice, stream);

    // one-time bf16 weight conversions
    cvt_wcomb_kernel<<<(2048 * 1024) / 256, 256, 0, stream>>>(W_ih, W_hh, wcomb);
    cvt_wfc_kernel<<<(32000 * 512 + 255) / 256, 256, 0, stream>>>(W_fc, wfcb, 32000 * 512);

    // sequential recurrence (stream order enforces dependencies)
    for (int t = 0; t < 64; ++t) {
        embed_gather_kernel<<<128, 256, 0, stream>>>(tokens, t, emb, hstate, apanel);
        gates_kernel<<<dim3(2, 16), 256, 0, stream>>>(apanel, wcomb, gates);
        lstm_update_kernel<<<64, 256, 0, stream>>>(gates, b_ih, b_hh, tokens, t,
                                                   hstate, cstate, hbf);
    }

    // batched vocab projection over all 64*32 = 2048 rows
    proj_kernel<<<dim3(128, 63), 256, 0, stream>>>(hbf, wfcb, b_fc, out);
}
